// AMK_Block_Old_36240934043648
// MI455X (gfx1250) — matrix operand order repaired and checked by reading, not compiled, not run on hardware
//
#include <hip/hip_runtime.h>
#include <hip/hip_bf16.h>
#include <math.h>

// ---------------------------------------------------------------------------
// Problem constants (fixed by the reference harness)
// ---------------------------------------------------------------------------
#define D_MODEL   1024
#define NUM_HEADS 16
#define HEAD_DIM  64
#define INNER     2816
#define NSEQ      2048
#define NBATCH    2
#define NROWS     (NBATCH * NSEQ)        // 4096
#define RMS_EPS   1.1920929e-07f

typedef __attribute__((ext_vector_type(16))) __bf16 bhalf16;
typedef __attribute__((ext_vector_type(8)))  __bf16 bhalf8;
typedef __attribute__((ext_vector_type(8)))  float  floatx8;

// CDNA5 async global->LDS copy (ASYNCcnt path), guarded so the file still
// compiles if the toolchain lacks the builtin. Parameter types per hipcc
// diagnostic: (v4i AS1*, v4i AS3*, imm int, imm int).
#if defined(__HIP_DEVICE_COMPILE__) && __has_builtin(__builtin_amdgcn_global_load_async_to_lds_b128)
#define HAVE_ASYNC 1
typedef int v4i_t __attribute__((vector_size(16)));
typedef __attribute__((address_space(1))) v4i_t g_v4i;
typedef __attribute__((address_space(3))) v4i_t l_v4i;
#else
#define HAVE_ASYNC 0
#endif

__device__ __forceinline__ void wait_async0() {
#if HAVE_ASYNC
#if __has_builtin(__builtin_amdgcn_s_wait_asynccnt)
  __builtin_amdgcn_s_wait_asynccnt(0);
#else
  asm volatile("s_wait_asynccnt 0x0" ::: "memory");
#endif
#endif
}

// ---------------------------------------------------------------------------
// bf16 <-> f32 via explicit bit manipulation (RNE), storage is unsigned short
// ---------------------------------------------------------------------------
__device__ __forceinline__ unsigned short f2bf(float f) {
  union { float f; unsigned u; } v; v.f = f;
  unsigned u = v.u;
  u += 0x7FFFu + ((u >> 16) & 1u);
  return (unsigned short)(u >> 16);
}
__device__ __forceinline__ float bf2f(unsigned short h) {
  union { unsigned u; float f; } v; v.u = ((unsigned)h) << 16;
  return v.f;
}
__device__ __forceinline__ float silu_f(float x) { return x / (1.f + __expf(-x)); }

__device__ __forceinline__ floatx8 wmma_bf16(bhalf16 a, bhalf16 b, floatx8 c) {
  return __builtin_amdgcn_wmma_f32_16x16x32_bf16(false, a, false, b, (short)0, c,
                                                 false, false);
}

// A-fragment (16-bit A 16x32): rowPtr = start of this row's 32-half
// K chunk in LDS. lane<16: K 0..15 ; lane>=16: K 16..31.
__device__ __forceinline__ bhalf16 load_afrag(const unsigned short* rowPtr, int laneHi) {
  return *(const bhalf16*)(rowPtr + laneHi * 16);
}
// B-fragment: colPtr = start of this column's 32-half K chunk (K-contiguous,
// i.e. B stored transposed in LDS). lane<16: K 0..15 ; lane>=16: K 16..31.
__device__ __forceinline__ bhalf16 load_bfrag(const unsigned short* colPtr, int laneHi) {
  return *(const bhalf16*)(colPtr + laneHi * 16);
}

// ---------------------------------------------------------------------------
// Generic bf16 GEMM:  out = epilogue( A[MxK] @ W[KxN] )
// Block tile 128(M) x 128(N), K-step 32, double-buffered LDS; 8 waves, each a
// 32x64 tile via 8 WMMAs per K-step. A tile staged with async global->LDS
// copies (ASYNCcnt) when available; B tile transposed through registers.
// MODE 0: f32 store            MODE 1: f32 accumulate (+=)
// MODE 2: bf16 store           MODE 3: phi=elu+1, bf16 -> (B,H,N,dh) layout
// MODE 4: bf16 -> (B,H,N,dh) layout
// ---------------------------------------------------------------------------
template <int MODE>
__global__ __launch_bounds__(256) void gemm_bf16_kernel(
    const unsigned short* __restrict__ A, const unsigned short* __restrict__ W,
    int M, int N, int K, float* __restrict__ outF,
    unsigned short* __restrict__ outB) {
  __shared__ __align__(32) unsigned short sA[2][128 * 32];  // [m][k]
  __shared__ __align__(32) unsigned short sB[2][128 * 32];  // [n][k] (transposed)

  const int tid = threadIdx.x;
  const int wave = tid >> 5, lane = tid & 31;
  const int lane16 = lane & 15, laneHi = lane >> 4;
  const int wm = wave >> 1, wn = wave & 1;          // 4 x 2 wave grid
  const int m0 = blockIdx.y * 128, n0 = blockIdx.x * 128;

  const int arow = tid >> 1, aseg = (tid & 1) * 16; // A: 2 threads/row, 32B each
  const int bkr = tid >> 4, bns = (tid & 15) * 8;   // B: 16 threads/k-row

  floatx8 z = {};
  floatx8 acc[2][4];
#pragma unroll
  for (int i = 0; i < 2; ++i)
#pragma unroll
    for (int j = 0; j < 4; ++j) acc[i][j] = z;

  const int nsteps = K / 32;

  auto stage = [&](int k0, int buf) {
    // A tile 128x32 row-major
#if HAVE_ASYNC
    {
      const unsigned short* g = A + (size_t)(m0 + arow) * K + k0 + aseg;
      unsigned short* l = &sA[buf][arow * 32 + aseg];
      __builtin_amdgcn_global_load_async_to_lds_b128((g_v4i*)g, (l_v4i*)l, 0, 0);
      __builtin_amdgcn_global_load_async_to_lds_b128((g_v4i*)(g + 8), (l_v4i*)(l + 8), 0, 0);
    }
#else
    {
      const uint4* g = (const uint4*)(A + (size_t)(m0 + arow) * K + k0 + aseg);
      uint4* s = (uint4*)(&sA[buf][arow * 32 + aseg]);
      s[0] = g[0]; s[1] = g[1];
    }
#endif
    // B tile 32(k) x 128(n), stored transposed -> sB[n][k]
#pragma unroll
    for (int j2 = 0; j2 < 2; ++j2) {
      int kr = bkr + j2 * 16;
      uint4 t = *(const uint4*)(W + (size_t)(k0 + kr) * N + n0 + bns);
      const unsigned short* p = (const unsigned short*)&t;
#pragma unroll
      for (int j = 0; j < 8; ++j) sB[buf][(bns + j) * 32 + kr] = p[j];
    }
  };

  stage(0, 0);

  for (int s = 0; s < nsteps; ++s) {
    const int buf = s & 1;
    wait_async0();
    __syncthreads();
    if (s + 1 < nsteps) stage((s + 1) * 32, buf ^ 1);

    bhalf16 aF[2], bF[4];
#pragma unroll
    for (int i = 0; i < 2; ++i)
      aF[i] = load_afrag(&sA[buf][(wm * 32 + i * 16 + lane16) * 32], laneHi);
#pragma unroll
    for (int j = 0; j < 4; ++j)
      bF[j] = load_bfrag(&sB[buf][(wn * 64 + j * 16 + lane16) * 32], laneHi);
#pragma unroll
    for (int i = 0; i < 2; ++i)
#pragma unroll
      for (int j = 0; j < 4; ++j)
        acc[i][j] = wmma_bf16(aF[i], bF[j], acc[i][j]);
  }

#pragma unroll
  for (int i = 0; i < 2; ++i)
#pragma unroll
    for (int j = 0; j < 4; ++j)
#pragma unroll
      for (int r = 0; r < 8; ++r) {
        int gm = m0 + wm * 32 + i * 16 + laneHi * 8 + r;
        int gn = n0 + wn * 64 + j * 16 + lane16;
        float v = acc[i][j][r];
        if constexpr (MODE == 0) {
          outF[(size_t)gm * N + gn] = v;
        } else if constexpr (MODE == 1) {
          outF[(size_t)gm * N + gn] += v;
        } else if constexpr (MODE == 2) {
          outB[(size_t)gm * N + gn] = f2bf(v);
        } else {
          int b = gm >> 11, n = gm & (NSEQ - 1);   // NSEQ = 2048
          int h = gn >> 6, d = gn & 63;
          size_t idx = (((size_t)(b * NUM_HEADS + h)) * NSEQ + n) * HEAD_DIM + d;
          if constexpr (MODE == 3) v = (v > 0.f) ? (v + 1.f) : __expf(v);  // elu+1
          outB[idx] = f2bf(v);
        }
      }
}

// ---------------------------------------------------------------------------
// Linear attention: per (b,h), 128 q-rows per block (8 waves x 16 rows).
// W = relu(phi_q . phi_k * dh^-0.5)^2 ; C = (W @ v) / (rowsum(W)+1e-6)
// ---------------------------------------------------------------------------
__global__ __launch_bounds__(256) void attn_kernel(
    const unsigned short* __restrict__ phiQ,
    const unsigned short* __restrict__ phiK,
    const unsigned short* __restrict__ Vh, float* __restrict__ Cout) {
  __shared__ __align__(32) unsigned short sQ[128 * 64];
  __shared__ __align__(32) unsigned short sK[32 * 64];
  __shared__ __align__(32) unsigned short sVT[64 * 32];   // transposed v
  __shared__ __align__(32) unsigned short sW[8][16 * 32]; // per-wave W staging

  const int tid = threadIdx.x;
  const int wave = tid >> 5, lane = tid & 31;
  const int lane16 = lane & 15, laneHi = lane >> 4;
  const int bh = blockIdx.x;
  const int q0 = blockIdx.y * 128;
  const size_t base = (size_t)bh * NSEQ * HEAD_DIM;

  {   // load phi_q block 128x64
    int row = tid >> 1, seg = (tid & 1) * 32;
    const uint4* g = (const uint4*)(phiQ + base + (size_t)(q0 + row) * 64 + seg);
    uint4* s = (uint4*)(sQ + row * 64 + seg);
    s[0] = g[0]; s[1] = g[1]; s[2] = g[2]; s[3] = g[3];
  }
  __syncthreads();

  bhalf16 aq[2];
#pragma unroll
  for (int ks = 0; ks < 2; ++ks)
    aq[ks] = load_afrag(sQ + (wave * 16 + lane16) * 64 + ks * 32, laneHi);

  floatx8 z = {};
  floatx8 acc[4]; acc[0] = z; acc[1] = z; acc[2] = z; acc[3] = z;
  float rsum[8];
#pragma unroll
  for (int r = 0; r < 8; ++r) rsum[r] = 0.f;

  const int lrow = tid >> 3;        // 0..31 key row
  const int lseg = (tid & 7) * 8;   // halves along dh

  for (int kc = 0; kc < NSEQ; kc += 32) {
    {   // phi_k chunk 32x64 row-major
      const uint4* g = (const uint4*)(phiK + base + (size_t)(kc + lrow) * 64 + lseg);
      *(uint4*)(sK + lrow * 64 + lseg) = g[0];
    }
    {   // v chunk transposed -> sVT[d][key]
      uint4 t = *(const uint4*)(Vh + base + (size_t)(kc + lrow) * 64 + lseg);
      const unsigned short* p = (const unsigned short*)&t;
#pragma unroll
      for (int j = 0; j < 8; ++j) sVT[(lseg + j) * 32 + lrow] = p[j];
    }
    __syncthreads();

    unsigned short* w = sW[wave];
#pragma unroll
    for (int kt = 0; kt < 2; ++kt) {    // two 16-key subtiles
      floatx8 wacc = z;
#pragma unroll
      for (int ks = 0; ks < 2; ++ks) {  // K = dh = 64 -> 2 steps of 32
        bhalf16 bk = load_bfrag(sK + (kt * 16 + lane16) * 64 + ks * 32, laneHi);
        wacc = wmma_bf16(aq[ks], bk, wacc);
      }
#pragma unroll
      for (int r = 0; r < 8; ++r) {
        float x = wacc[r] * 0.125f;     // dh^-0.5
        x = (x > 0.f) ? x : 0.f;
        x = x * x;                      // KPOW = 2
        rsum[r] += x;
        w[(laneHi * 8 + r) * 32 + kt * 16 + lane16] = f2bf(x);
      }
    }
    // W(16x32keys) @ v(32keys x 64dh)
    bhalf16 aW = load_afrag(w + lane16 * 32, laneHi);
#pragma unroll
    for (int nt = 0; nt < 4; ++nt) {
      bhalf16 bv = load_bfrag(sVT + (nt * 16 + lane16) * 32, laneHi);
      acc[nt] = wmma_bf16(aW, bv, acc[nt]);
    }
    __syncthreads();
  }

  // rowsum: reduce across the 16 lanes holding key-columns (halves stay apart)
#pragma unroll
  for (int d = 1; d < 16; d <<= 1)
#pragma unroll
    for (int r = 0; r < 8; ++r) rsum[r] += __shfl_xor(rsum[r], d, 32);

#pragma unroll
  for (int nt = 0; nt < 4; ++nt)
#pragma unroll
    for (int r = 0; r < 8; ++r) {
      int row = q0 + wave * 16 + laneHi * 8 + r;
      int d = nt * 16 + lane16;
      Cout[base + (size_t)row * 64 + d] = acc[nt][r] / (rsum[r] + 1e-6f);
    }
}

// ---------------------------------------------------------------------------
// Small elementwise kernels
// ---------------------------------------------------------------------------
__global__ void f2bf_kernel(const float* __restrict__ in,
                            unsigned short* __restrict__ out, size_t n) {
  size_t i = (size_t)blockIdx.x * 256 + threadIdx.x;
  if (i < n) out[i] = f2bf(in[i]);
}

__global__ void addbf_kernel(const float* __restrict__ a, const float* __restrict__ b,
                             unsigned short* __restrict__ out, size_t n) {
  size_t i = (size_t)blockIdx.x * 256 + threadIdx.x;
  if (i < n) out[i] = f2bf(a[i] + b[i]);
}

// m = C - v, scatter both from (B,H,N,dh) to (B,N,D) bf16
__global__ void mc_kernel(const float* __restrict__ C,
                          const unsigned short* __restrict__ Vh,
                          unsigned short* __restrict__ mOut,
                          unsigned short* __restrict__ cOut) {
  size_t i = (size_t)blockIdx.x * 256 + threadIdx.x;  // over B*H*N*dh
  int d = (int)(i & 63);
  int n = (int)((i >> 6) & (NSEQ - 1));
  int bh = (int)(i >> 17);
  int b = bh >> 4, h = bh & 15;
  float c = C[i];
  float m = c - bf2f(Vh[i]);
  size_t dst = ((size_t)(b * NSEQ + n)) * D_MODEL + h * HEAD_DIM + d;
  mOut[dst] = f2bf(m);
  cOut[dst] = f2bf(c);
}

// out = rmsnorm(X0 + X1, g); one block per row of D_MODEL
__global__ __launch_bounds__(256) void rmsnorm_kernel(
    const float* __restrict__ X0, const float* __restrict__ X1,
    const float* __restrict__ g, float* __restrict__ outF,
    unsigned short* __restrict__ outB) {
  __shared__ float red[256];
  const size_t base = (size_t)blockIdx.x * D_MODEL;
  const int tid = threadIdx.x;
  float xs[4];
  float ss = 0.f;
#pragma unroll
  for (int j = 0; j < 4; ++j) {
    int c = tid + j * 256;
    float x = X0[base + c] + X1[base + c];
    xs[j] = x;
    ss += x * x;
  }
  red[tid] = ss;
  __syncthreads();
  for (int s = 128; s > 0; s >>= 1) {
    if (tid < s) red[tid] += red[tid + s];
    __syncthreads();
  }
  float r = rsqrtf(red[0] / (float)D_MODEL + RMS_EPS);
#pragma unroll
  for (int j = 0; j < 4; ++j) {
    int c = tid + j * 256;
    float y = xs[j] * r * g[c];
    if (outF) outF[base + c] = y;
    if (outB) outB[base + c] = f2bf(y);
  }
}

// Hf = silu(G) * U  from GU[NROWS][2*INNER]
__global__ void hf_kernel(const unsigned short* __restrict__ GU,
                          unsigned short* __restrict__ Hf) {
  size_t i = (size_t)blockIdx.x * 256 + threadIdx.x;
  if (i >= (size_t)NROWS * INNER) return;
  size_t row = i / INNER;
  int ii = (int)(i - row * INNER);
  float g = bf2f(GU[row * (2 * INNER) + ii]);
  float u = bf2f(GU[row * (2 * INNER) + INNER + ii]);
  Hf[i] = f2bf(silu_f(g) * u);
}

// depthwise width-3 conv along n, + bias, silu
__global__ void conv_kernel(const unsigned short* __restrict__ Hf,
                            const float* __restrict__ cw, const float* __restrict__ cb,
                            unsigned short* __restrict__ Hc) {
  size_t i = (size_t)blockIdx.x * 256 + threadIdx.x;
  if (i >= (size_t)NROWS * INNER) return;
  size_t row = i / INNER;
  int ii = (int)(i - row * INNER);
  int n = (int)(row & (NSEQ - 1));
  float xm = (n > 0)        ? bf2f(Hf[i - INNER]) : 0.f;
  float x0 = bf2f(Hf[i]);
  float xp = (n < NSEQ - 1) ? bf2f(Hf[i + INNER]) : 0.f;
  float y = cw[ii * 3 + 0] * xm + cw[ii * 3 + 1] * x0 + cw[ii * 3 + 2] * xp + cb[ii];
  Hc[i] = f2bf(silu_f(y));
}

// ---------------------------------------------------------------------------
// Host-side orchestration
// ---------------------------------------------------------------------------
extern "C" void kernel_launch(void* const* d_in, const int* in_sizes, int n_in,
                              void* d_out, int out_size, void* d_ws, size_t ws_size,
                              hipStream_t stream) {
  const float* Qin    = (const float*)d_in[0];
  const float* X      = (const float*)d_in[1];
  const float* wq     = (const float*)d_in[2];
  const float* wk     = (const float*)d_in[3];
  const float* wv     = (const float*)d_in[4];
  const float* wo     = (const float*)d_in[5];
  const float* waux   = (const float*)d_in[6];
  const float* w_up   = (const float*)d_in[7];
  const float* conv_w = (const float*)d_in[8];
  const float* conv_b = (const float*)d_in[9];
  const float* w_down = (const float*)d_in[10];
  const float* g1     = (const float*)d_in[11];
  const float* g2     = (const float*)d_in[12];
  float* out = (float*)d_out;

  // workspace carve-up (~242 MB total)
  char* ws = (char*)d_ws;
  size_t off = 0;
  auto alloc = [&](size_t bytes) { void* p = ws + off; off += (bytes + 255) & ~(size_t)255; return p; };
  unsigned short* wq_b    = (unsigned short*)alloc((size_t)D_MODEL * D_MODEL * 2);
  unsigned short* wk_b    = (unsigned short*)alloc((size_t)D_MODEL * D_MODEL * 2);
  unsigned short* wv_b    = (unsigned short*)alloc((size_t)D_MODEL * D_MODEL * 2);
  unsigned short* wo_b    = (unsigned short*)alloc((size_t)D_MODEL * D_MODEL * 2);
  unsigned short* waux_b  = (unsigned short*)alloc((size_t)D_MODEL * D_MODEL * 2);
  unsigned short* wup_b   = (unsigned short*)alloc((size_t)D_MODEL * 2 * INNER * 2);
  unsigned short* wdown_b = (unsigned short*)alloc((size_t)INNER * D_MODEL * 2);
  unsigned short* Hc_b    = (unsigned short*)alloc((size_t)NROWS * D_MODEL * 2);
  unsigned short* phiq_b  = (unsigned short*)alloc((size_t)NROWS * D_MODEL * 2);
  unsigned short* phik_b  = (unsigned short*)alloc((size_t)NROWS * D_MODEL * 2);
  unsigned short* v_b     = (unsigned short*)alloc((size_t)NROWS * D_MODEL * 2);
  float*          C_f     = (float*)alloc((size_t)NROWS * D_MODEL * 4);
  unsigned short* m_b     = (unsigned short*)alloc((size_t)NROWS * D_MODEL * 2);
  unsigned short* Cc_b    = (unsigned short*)alloc((size_t)NROWS * D_MODEL * 2);
  float*          mproj_f = (float*)alloc((size_t)NROWS * D_MODEL * 4);
  float*          Qi_f    = (float*)alloc((size_t)NROWS * D_MODEL * 4);
  unsigned short* Qi_b    = (unsigned short*)alloc((size_t)NROWS * D_MODEL * 2);
  unsigned short* GU_b    = (unsigned short*)alloc((size_t)NROWS * 2 * INNER * 2);
  unsigned short* Hf_b    = (unsigned short*)alloc((size_t)NROWS * INNER * 2);
  unsigned short* Hcv_b   = (unsigned short*)alloc((size_t)NROWS * INNER * 2);
  float*          Hout_f  = (float*)alloc((size_t)NROWS * D_MODEL * 4);
  (void)ws_size; (void)in_sizes; (void)n_in; (void)out_size;

  const int T = 256;
  auto blocks = [](size_t n) { return (unsigned)((n + 255) / 256); };

  // 1. weights -> bf16
  f2bf_kernel<<<blocks((size_t)D_MODEL * D_MODEL), T, 0, stream>>>(wq, wq_b, (size_t)D_MODEL * D_MODEL);
  f2bf_kernel<<<blocks((size_t)D_MODEL * D_MODEL), T, 0, stream>>>(wk, wk_b, (size_t)D_MODEL * D_MODEL);
  f2bf_kernel<<<blocks((size_t)D_MODEL * D_MODEL), T, 0, stream>>>(wv, wv_b, (size_t)D_MODEL * D_MODEL);
  f2bf_kernel<<<blocks((size_t)D_MODEL * D_MODEL), T, 0, stream>>>(wo, wo_b, (size_t)D_MODEL * D_MODEL);
  f2bf_kernel<<<blocks((size_t)D_MODEL * D_MODEL), T, 0, stream>>>(waux, waux_b, (size_t)D_MODEL * D_MODEL);
  f2bf_kernel<<<blocks((size_t)D_MODEL * 2 * INNER), T, 0, stream>>>(w_up, wup_b, (size_t)D_MODEL * 2 * INNER);
  f2bf_kernel<<<blocks((size_t)INNER * D_MODEL), T, 0, stream>>>(w_down, wdown_b, (size_t)INNER * D_MODEL);

  // 2. Hc = Q_in + X -> bf16
  addbf_kernel<<<blocks((size_t)NROWS * D_MODEL), T, 0, stream>>>(Qin, X, Hc_b, (size_t)NROWS * D_MODEL);

  // 3. q, k, v projections (WMMA GEMMs). MODE 3 = phi+head layout, 4 = head layout.
  dim3 gqkv(D_MODEL / 128, NROWS / 128);
  gemm_bf16_kernel<3><<<gqkv, T, 0, stream>>>(Hc_b, wq_b, NROWS, D_MODEL, D_MODEL, nullptr, phiq_b);
  gemm_bf16_kernel<3><<<gqkv, T, 0, stream>>>(Hc_b, wk_b, NROWS, D_MODEL, D_MODEL, nullptr, phik_b);
  gemm_bf16_kernel<4><<<gqkv, T, 0, stream>>>(Hc_b, wv_b, NROWS, D_MODEL, D_MODEL, nullptr, v_b);

  // 4. linear attention -> C (B,H,N,dh) f32
  dim3 gattn(NBATCH * NUM_HEADS, NSEQ / 128);
  attn_kernel<<<gattn, T, 0, stream>>>(phiq_b, phik_b, v_b, C_f);

  // 5. m = C - v ; transpose both to (B,N,D) bf16
  mc_kernel<<<blocks((size_t)NROWS * D_MODEL), T, 0, stream>>>(C_f, v_b, m_b, Cc_b);

  // 6. m_proj = m @ wo + C @ waux  (f32 store then f32 accumulate)
  gemm_bf16_kernel<0><<<gqkv, T, 0, stream>>>(m_b, wo_b, NROWS, D_MODEL, D_MODEL, mproj_f, nullptr);
  gemm_bf16_kernel<1><<<gqkv, T, 0, stream>>>(Cc_b, waux_b, NROWS, D_MODEL, D_MODEL, mproj_f, nullptr);

  // 7. Qi = rmsnorm(Q_in + m_proj, g1) -> f32 + bf16
  rmsnorm_kernel<<<NROWS, T, 0, stream>>>(Qin, mproj_f, g1, Qi_f, Qi_b);

  // 8. GU = Qi @ w_up -> bf16
  dim3 gup((2 * INNER) / 128, NROWS / 128);
  gemm_bf16_kernel<2><<<gup, T, 0, stream>>>(Qi_b, wup_b, NROWS, 2 * INNER, D_MODEL, nullptr, GU_b);

  // 9. Hf = silu(G) * U
  hf_kernel<<<blocks((size_t)NROWS * INNER), T, 0, stream>>>(GU_b, Hf_b);

  // 10. depthwise conv + bias + silu
  conv_kernel<<<blocks((size_t)NROWS * INNER), T, 0, stream>>>(Hf_b, conv_w, conv_b, Hcv_b);

  // 11. H_out = H_conv @ w_down -> f32
  dim3 gdown(D_MODEL / 128, NROWS / 128);
  gemm_bf16_kernel<0><<<gdown, T, 0, stream>>>(Hcv_b, wdown_b, NROWS, D_MODEL, INNER, Hout_f, nullptr);

  // 12. out = rmsnorm(Qi + H_out, g2) -> f32 d_out
  rmsnorm_kernel<<<NROWS, T, 0, stream>>>(Qi_f, Hout_f, g2, out, nullptr);
}
